// IMDCTSymExpHead_2310692405898
// MI455X (gfx1250) — matrix operand order repaired and checked by reading, not compiled, not run on hardware
//
#include <hip/hip_runtime.h>
#include <hip/hip_bf16.h>
#include <math.h>

// ---------------------------------------------------------------------------
// Problem constants (from reference)
// ---------------------------------------------------------------------------
constexpr int Bc   = 8;
constexpr int DIM  = 512;
constexpr int GIN  = 256;
constexpr int Lc   = 4096;
constexpr int Nc   = 256;    // mdct frame len / out channels of pconv2
constexpr int Rr   = 2;
constexpr int Kc   = 5;
constexpr int D2   = 2 * DIM;      // 1024
constexpr int N2   = 2 * Nc;       // 512 (imdct frame, 2N)

typedef __attribute__((ext_vector_type(16))) _Float16     v16h;
typedef __attribute__((ext_vector_type(8)))  _Float16     v8h;
typedef __attribute__((ext_vector_type(8)))  float        v8f;
typedef __attribute__((ext_vector_type(4)))  unsigned int v4u;
typedef __attribute__((ext_vector_type(8)))  int          v8i;
typedef __attribute__((ext_vector_type(4)))  int          v4i;

static __device__ __forceinline__ v16h ld_a_frag(const _Float16* p0, const _Float16* p1) {
    // A-fragment per lane: two contiguous 8xf16 (16B) chunks (K = h*8.., K = 16+h*8..)
    v8h a = *(const v8h*)p0;
    v8h b = *(const v8h*)p1;
    return __builtin_shufflevector(a, b, 0,1,2,3,4,5,6,7,8,9,10,11,12,13,14,15);
}

// ---------------------------------------------------------------------------
// 1) Adapter GEMV:  out[b,o] = ain_w[o,:] . g[b,:] + ain_b[o]
// ---------------------------------------------------------------------------
__global__ __launch_bounds__(256)
void adapter_gemv(const float* __restrict__ w, const float* __restrict__ bias,
                  const float* __restrict__ g, float* __restrict__ out, int rows) {
    int idx = blockIdx.x * 256 + threadIdx.x;
    if (idx >= Bc * rows) return;
    int o = idx % rows;
    int b = idx / rows;
    const float* gw = g + b * GIN;
    const float* wr = w + (long long)o * GIN;
    float acc = bias[o];
    #pragma unroll 8
    for (int c = 0; c < GIN; ++c) acc += wr[c] * gw[c];
    out[idx] = acc;
}

// ---------------------------------------------------------------------------
// 2) Fold rank-2 LoRA into per-batch effective weight (f16)
//    weff[b,co,c] = w[co,c] + sum_r aout[b, r*cout+co] * ain[b, c*R + r]
// ---------------------------------------------------------------------------
__global__ __launch_bounds__(256)
void weff_build(const float* __restrict__ w, const float* __restrict__ ain,
                const float* __restrict__ aout, _Float16* __restrict__ weff,
                int cout, int cin) {
    long long idx = (long long)blockIdx.x * 256 + threadIdx.x;
    if (idx >= (long long)Bc * cout * cin) return;
    int c  = (int)(idx % cin);
    long long rest = idx / cin;
    int co = (int)(rest % cout);
    int b  = (int)(rest / cout);
    const float* ai = ain  + (long long)b * cin * Rr;
    const float* ao = aout + (long long)b * Rr * cout;
    float d = ao[co] * ai[c * Rr + 0] + ao[cout + co] * ai[c * Rr + 1];
    weff[idx] = (_Float16)(w[(long long)co * cin + c] + d);
}

// ---------------------------------------------------------------------------
// 3) Depthwise causal conv + transpose to time-major f16: h0t[b,l,c]
// ---------------------------------------------------------------------------
__global__ __launch_bounds__(256)
void dwconv_transpose(const float* __restrict__ x, const float* __restrict__ w,
                      const float* __restrict__ bias, _Float16* __restrict__ h0t) {
    long long idx = (long long)blockIdx.x * 256 + threadIdx.x;   // over B*L*DIM
    int c = (int)(idx & (DIM - 1));
    int l = (int)((idx >> 9) & (Lc - 1));
    int b = (int)(idx >> 21);
    const float* xr = x + ((long long)b * DIM + c) * Lc;
    float acc = bias[c];
    #pragma unroll
    for (int j = 0; j < Kc; ++j)
        if (l >= j) acc += w[c * Kc + j] * xr[l - j];
    h0t[((long long)b * Lc + l) * DIM + c] = (_Float16)acc;
}

// ---------------------------------------------------------------------------
// 4) Windowed MDCT-IV synthesis basis (2N x N), f16, row-major
// ---------------------------------------------------------------------------
__global__ __launch_bounds__(256)
void imdct_basis(_Float16* __restrict__ bas) {
    int idx = blockIdx.x * 256 + threadIdx.x;          // over 512*256
    if (idx >= N2 * Nc) return;
    int m = idx / Nc;
    int n = idx % Nc;
    double n0  = (Nc + 1) * 0.5;
    double arg = M_PI * ((double)m + n0) * (2.0 * n + 1.0) / (double)N2;
    double win = sin(M_PI * ((double)m + 0.5) / (double)N2);
    double s   = sqrt(2.0 / (double)Nc);
    bas[idx] = (_Float16)(s * cos(arg) * win);
}

// ---------------------------------------------------------------------------
// 5) WMMA GEMM over time-major activations, B tile staged in LDS via TDM.
//    Y[b,l,m] = epi( sum_k A[b,m,k] * X[b,l,k] + bias[m] )
//    Block = 256 threads = 8 waves; all waves share one 16-frame B tile in
//    LDS (loaded once by TDM, TENSORcnt-tracked); each wave owns MFRAG 16x16
//    output tiles stacked in M. A fragments use a copy-free ping-pong double
//    buffer (K stepped by 64).  LDS tile is padded (4 dwords per 128/256-dword
//    chunk) so fragment ds_load_b128s are bank-conflict-free (2-way worst
//    case for CIN=1024).
// ---------------------------------------------------------------------------
template<int CIN, int MFRAG, int EPI>
__global__ __launch_bounds__(256)
void gemm_wmma(const _Float16* __restrict__ A, long long aStride,
               const _Float16* __restrict__ X, const float* __restrict__ bias,
               _Float16* __restrict__ Y, int cout) {
    constexpr int CH_SH  = (CIN == 256) ? 8 : 9;     // log2 halves per pad chunk
    constexpr int PAD_IV = (CIN == 256) ? 6 : 7;     // TDM pad_interval encoding
    constexpr int ROWH   = CIN + (CIN >> CH_SH) * 8; // padded halves per row

    __shared__ alignas(32) _Float16 smem[16 * ROWH];

    const int lane = threadIdx.x & 31;
    const int wave = threadIdx.x >> 5;
    const int r    = lane & 15;
    const int h    = lane >> 4;
    const int b    = blockIdx.z;
    const int colb = blockIdx.x * 16;
    const int mbase = blockIdx.y * (8 * MFRAG * 16) + wave * (MFRAG * 16);

    const _Float16* Ab      = A + (long long)b * aStride;
    const _Float16* tileSrc = X + ((long long)(b * Lc + colb)) * CIN;

#if __has_builtin(__builtin_amdgcn_tensor_load_to_lds) && __has_builtin(__builtin_amdgcn_s_wait_tensorcnt)
    if (wave == 0) {
        // ---- Tensor DMA descriptor (D#), 2D tile: 16 rows x CIN halves ----
        unsigned long long ga = (unsigned long long)(const void*)tileSrc;
        unsigned int ldsOff   = (unsigned int)(unsigned long long)(const void*)&smem[0];
        v4u g0;
        g0[0] = 1u;                                            // count=1, user mode
        g0[1] = ldsOff;                                        // lds_addr
        g0[2] = (unsigned int)ga;                              // global_addr[31:0]
        g0[3] = (unsigned int)((ga >> 32) & 0x01FFFFFFull)     // global_addr[56:32]
              | (2u << 30);                                    // type = image
        v8i g1;
        g1[0] = (1 << 16)                                      // data_size = 2B
              | (1 << 20)                                      // pad_enable
              | (PAD_IV << 22)                                 // pad_interval
              | (3 << 25);                                     // pad_amount = 4 dwords
        g1[1] = (CIN & 0xFFFF) << 16;                          // tensor_dim0[15:0]
        g1[2] = ((CIN >> 16) & 0xFFFF) | (16 << 16);           // td0[31:16] | tensor_dim1[15:0]
        g1[3] = (CIN << 16);                                   // td1[31:16]=0 | tile_dim0
        g1[4] = 16;                                            // tile_dim1=16, tile_dim2=0
        g1[5] = CIN;                                           // tensor_dim0_stride[31:0]
        g1[6] = ((16 * CIN) & 0xFFFF) << 16;                   // s0[47:32]=0 | s1[15:0]
        g1[7] = (16 * CIN) >> 16;                              // tensor_dim1_stride[47:16]
        v4i g2 = {0, 0, 0, 0};                                 // unused (<=2D tensor)
        v4i g3 = {0, 0, 0, 0};                                 // unused (<=2D tensor)
        v8i g4 = {0, 0, 0, 0, 0, 0, 0, 0};                     // extra group (clang-23 arity)
        __builtin_amdgcn_tensor_load_to_lds(g0, g1, g2, g3, g4, 0);
        __builtin_amdgcn_s_wait_tensorcnt(0);
    }
#else
    // Fallback: cooperative global->LDS copy with identical pad mapping.
    for (int i = threadIdx.x; i < 16 * (CIN / 8); i += 256) {
        int row = i / (CIN / 8);
        int k   = (i % (CIN / 8)) * 8;
        v8h v = *(const v8h*)(tileSrc + (long long)row * CIN + k);
        int d = row * CIN + k;
        *(v8h*)&smem[d + ((d >> CH_SH) << 3)] = v;
    }
#endif
    __syncthreads();

    v8f acc[MFRAG];
    #pragma unroll
    for (int f = 0; f < MFRAG; ++f) acc[f] = (v8f){0.f,0.f,0.f,0.f,0.f,0.f,0.f,0.f};

    const _Float16* arow[MFRAG];
    #pragma unroll
    for (int f = 0; f < MFRAG; ++f)
        arow[f] = Ab + (long long)(mbase + f * 16 + r) * CIN + h * 8;

    // Copy-free ping-pong K loop (CIN divisible by 64): bufA holds the even
    // 32-wide K step, bufB the odd one; loads land directly in WMMA operands.
    v16h bufA[MFRAG], bufB[MFRAG];
    #pragma unroll
    for (int f = 0; f < MFRAG; ++f) bufA[f] = ld_a_frag(arow[f], arow[f] + 16);

    for (int k0 = 0; k0 < CIN; k0 += 64) {
        if (k0 + 32 < CIN) {
            #pragma unroll
            for (int f = 0; f < MFRAG; ++f)
                bufB[f] = ld_a_frag(arow[f] + k0 + 32, arow[f] + k0 + 48);
        }
        {
            int d  = r * CIN + k0 + h * 8;
            int d2 = d + 16;
            v16h bf = ld_a_frag(&smem[d + ((d >> CH_SH) << 3)], &smem[d2 + ((d2 >> CH_SH) << 3)]);
            #pragma unroll
            for (int f = 0; f < MFRAG; ++f)
                acc[f] = __builtin_amdgcn_wmma_f32_16x16x32_f16(
                    false, bufA[f], false, bf, (short)0, acc[f], false, false);
        }
        if (k0 + 64 < CIN) {
            #pragma unroll
            for (int f = 0; f < MFRAG; ++f)
                bufA[f] = ld_a_frag(arow[f] + k0 + 64, arow[f] + k0 + 80);
        }
        {
            int d  = r * CIN + k0 + 32 + h * 8;
            int d2 = d + 16;
            v16h bf = ld_a_frag(&smem[d + ((d >> CH_SH) << 3)], &smem[d2 + ((d2 >> CH_SH) << 3)]);
            #pragma unroll
            for (int f = 0; f < MFRAG; ++f)
                acc[f] = __builtin_amdgcn_wmma_f32_16x16x32_f16(
                    false, bufB[f], false, bf, (short)0, acc[f], false, false);
        }
    }

    // Epilogue: D layout -> VGPR v holds (M = mtile + v + 8*h, Ncol = r)
    const long long yrow = ((long long)(b * Lc + colb + r)) * cout;
    #pragma unroll
    for (int f = 0; f < MFRAG; ++f) {
        union { v8f v; float s[8]; } u; u.v = acc[f];
        #pragma unroll
        for (int v = 0; v < 8; ++v) {
            int m = mbase + f * 16 + v + 8 * h;
            float val = u.s[v];
            if (bias) val += bias[m];
            if (EPI == 1) {                       // exact GELU
                val = 0.5f * val * (1.0f + erff(val * 0.70710678118654752f));
            } else if (EPI == 2) {                // symexp + clip
                float e = expf(fabsf(val)) - 1.0f;
                val = copysignf(e, val);
                val = fminf(100.0f, fmaxf(-100.0f, val));
            }
            Y[yrow + m] = (_Float16)val;
        }
    }
}

// ---------------------------------------------------------------------------
// 6) Overlap-add + 'same' crop: every output element written exactly once.
//    out[b,t'] = res[b, l, m] + res[b, l-1, m+N], with t = t'+N/2, l=t/N, m=t%N
// ---------------------------------------------------------------------------
__global__ __launch_bounds__(256)
void overlap_add(const _Float16* __restrict__ res, float* __restrict__ out) {
    long long idx = (long long)blockIdx.x * 256 + threadIdx.x;  // over B*L*N
    long long per = (long long)Lc * Nc;
    int b  = (int)(idx / per);
    long long tp = idx - (long long)b * per;
    long long t  = tp + Nc / 2;
    int l = (int)(t >> 8);          // /N (N=256)
    int m = (int)(t & (Nc - 1));
    float v = 0.0f;
    if (l < Lc)  v += (float)res[((long long)b * Lc + l) * N2 + m];
    if (l >= 1)  v += (float)res[((long long)b * Lc + (l - 1)) * N2 + m + Nc];
    out[idx] = v;
}

// ---------------------------------------------------------------------------
// Host launcher
// ---------------------------------------------------------------------------
extern "C" void kernel_launch(void* const* d_in, const int* in_sizes, int n_in,
                              void* d_out, int out_size, void* d_ws, size_t ws_size,
                              hipStream_t stream) {
    const float* x       = (const float*)d_in[0];
    const float* g       = (const float*)d_in[1];
    const float* dconv_w = (const float*)d_in[2];
    const float* dconv_b = (const float*)d_in[3];
    const float* w1      = (const float*)d_in[4];
    const float* b1      = (const float*)d_in[5];
    const float* ain1_w  = (const float*)d_in[6];
    const float* ain1_b  = (const float*)d_in[7];
    const float* aout1_w = (const float*)d_in[8];
    const float* aout1_b = (const float*)d_in[9];
    const float* w2      = (const float*)d_in[10];
    const float* b2      = (const float*)d_in[11];
    const float* ain2_w  = (const float*)d_in[12];
    const float* ain2_b  = (const float*)d_in[13];
    const float* aout2_w = (const float*)d_in[14];
    const float* aout2_b = (const float*)d_in[15];
    float* out = (float*)d_out;

    // Workspace layout (all 256B aligned); total ~156.4 MB
    char* ws = (char*)d_ws;
    size_t off = 0;
    auto alloc = [&](size_t bytes) { char* p = ws + off; off += (bytes + 255) & ~size_t(255); return p; };
    _Float16* h0t   = (_Float16*)alloc((size_t)Bc * Lc * DIM * 2);   // 32 MB
    _Float16* h1t   = (_Float16*)alloc((size_t)Bc * Lc * D2  * 2);   // 64 MB
    _Float16* xt    = (_Float16*)alloc((size_t)Bc * Lc * Nc  * 2);   // 16 MB
    _Float16* res   = (_Float16*)alloc((size_t)Bc * Lc * N2  * 2);   // 32 MB
    _Float16* weff1 = (_Float16*)alloc((size_t)Bc * D2 * DIM * 2);   // 8 MB
    _Float16* weff2 = (_Float16*)alloc((size_t)Bc * Nc * D2  * 2);   // 4 MB
    _Float16* bas   = (_Float16*)alloc((size_t)N2 * Nc * 2);         // 256 KB
    float* ain1 = (float*)alloc((size_t)Bc * DIM * Rr * 4);
    float* aout1= (float*)alloc((size_t)Bc * Rr * D2  * 4);
    float* ain2 = (float*)alloc((size_t)Bc * D2 * Rr  * 4);
    float* aout2= (float*)alloc((size_t)Bc * Rr * Nc  * 4);
    (void)ws_size; (void)in_sizes; (void)n_in; (void)out_size;

    // 1) LoRA adapter GEMVs
    adapter_gemv<<<(Bc*DIM*Rr)/256,  256, 0, stream>>>(ain1_w,  ain1_b,  g, ain1,  DIM*Rr);
    adapter_gemv<<<(Bc*Rr*D2)/256,   256, 0, stream>>>(aout1_w, aout1_b, g, aout1, Rr*D2);
    adapter_gemv<<<(Bc*D2*Rr)/256,   256, 0, stream>>>(ain2_w,  ain2_b,  g, ain2,  D2*Rr);
    adapter_gemv<<<(Bc*Rr*Nc)/256,   256, 0, stream>>>(aout2_w, aout2_b, g, aout2, Rr*Nc);

    // 2) Fold LoRA into per-batch effective weights (f16)
    weff_build<<<(Bc*D2*DIM)/256, 256, 0, stream>>>(w1, ain1, aout1, weff1, D2, DIM);
    weff_build<<<(Bc*Nc*D2)/256,  256, 0, stream>>>(w2, ain2, aout2, weff2, Nc, D2);

    // 3) Depthwise causal conv + transpose to time-major f16
    dwconv_transpose<<<(Bc*Lc*DIM)/256, 256, 0, stream>>>(x, dconv_w, dconv_b, h0t);

    // 4) IMDCT basis
    imdct_basis<<<(N2*Nc + 255)/256, 256, 0, stream>>>(bas);

    // 5) GEMM1 + GELU: h1t = gelu(Weff1[b] @ h0t + b1)   (1024 x 512 x 32768)
    {
        dim3 grid(Lc/16, D2/(8*4*16), Bc);               // (256, 2, 8)
        gemm_wmma<DIM,4,1><<<grid, 256, 0, stream>>>(weff1, (long long)D2*DIM, h0t, b1, h1t, D2);
    }
    // 6) GEMM2 + symexp + clip: xt = clip(symexp(Weff2[b] @ h1t + b2))  (256 x 1024)
    {
        dim3 grid(Lc/16, Nc/(8*2*16), Bc);               // (256, 1, 8)
        gemm_wmma<D2,2,2><<<grid, 256, 0, stream>>>(weff2, (long long)Nc*D2, h1t, b2, xt, Nc);
    }
    // 7) IMDCT GEMM: res = Bas @ xt   (512 x 256, shared A)
    {
        dim3 grid(Lc/16, N2/(8*4*16), Bc);               // (256, 1, 8)
        gemm_wmma<Nc,4,0><<<grid, 256, 0, stream>>>(bas, 0LL, xt, (const float*)nullptr, res, N2);
    }
    // 8) Overlap-add + crop -> d_out
    overlap_add<<<(Bc*Lc*Nc)/256, 256, 0, stream>>>(res, out);
}
